// Heavy_HitterLlamaAttention_79310866088656
// MI455X (gfx1250) — matrix operand order repaired and checked by reading, not compiled, not run on hardware
//
#include <hip/hip_runtime.h>

// ---------------------------------------------------------------------------
// Llama attention + H2O heavy-hitter scores, MI455X (gfx1250, wave32, WMMA)
// Pipeline: f32->bf16 convert (weights transposed to [N][K]) -> QKV GEMMs
//           (bf16 WMMA, f32 accum, async-LDS tile staging) -> RoPE ->
//           V transpose -> 2-pass flash attention (WMMA) w/ hh atomics ->
//           O GEMM with Wo into d_out.
// Requires ws_size >= ~227 MB.
// ---------------------------------------------------------------------------

typedef __attribute__((ext_vector_type(16))) __bf16        v16bf;
typedef __attribute__((ext_vector_type(8)))  float         v8f;
typedef __attribute__((ext_vector_type(8)))  unsigned int  v8u;

constexpr int SEQ = 2048;
constexpr int HID = 4096;
constexpr int NH  = 32;
constexpr int NKV = 8;
constexpr int HD  = 128;
constexpr int KVW = NKV * HD;              // 1024
constexpr float QK_SCALE = 0.08838834764831845f;   // 1/sqrt(128)
constexpr float ROPE_THETA = 500000.0f;

// ---- workspace layout (bytes) ----
constexpr size_t OFF_XB  = 0;                          // X bf16   [2048,4096]
constexpr size_t OFF_WQB = OFF_XB  + (size_t)SEQ*HID*2;        // Wq^T bf16 [4096,4096]
constexpr size_t OFF_WKB = OFF_WQB + (size_t)HID*HID*2;        // Wk^T bf16 [1024,4096]
constexpr size_t OFF_WVB = OFF_WKB + (size_t)HID*KVW*2;        // Wv^T bf16 [1024,4096]
constexpr size_t OFF_WOB = OFF_WVB + (size_t)HID*KVW*2;        // Wo^T bf16 [4096,4096]
constexpr size_t OFF_QF  = OFF_WOB + (size_t)HID*HID*2;        // Q f32
constexpr size_t OFF_KF  = OFF_QF  + (size_t)SEQ*HID*4;        // K f32
constexpr size_t OFF_VF  = OFF_KF  + (size_t)SEQ*KVW*4;        // V f32
constexpr size_t OFF_QB  = OFF_VF  + (size_t)SEQ*KVW*4;        // Q roped bf16
constexpr size_t OFF_KB  = OFF_QB  + (size_t)SEQ*HID*2;        // K roped bf16
constexpr size_t OFF_VT  = OFF_KB  + (size_t)SEQ*KVW*2;        // V^T bf16 [kv][d][t]
constexpr size_t OFF_ML  = OFF_VT  + (size_t)NKV*HD*SEQ*2;     // (m,l) float2 [head][row]
constexpr size_t OFF_OF  = OFF_ML  + (size_t)NH*SEQ*8;         // attn out f32
constexpr size_t OFF_OB  = OFF_OF  + (size_t)SEQ*HID*4;        // attn out bf16
// total ~227,016,704 bytes

__device__ __forceinline__ __bf16 f2bf(float f) { return (__bf16)f; }
// A-fragment (16x32 bf16) k index for reg r, lane-half h (even; pairs contiguous)
__device__ __forceinline__ int kidxA(int r, int h) { return 2*r + 8*h + (r >= 4 ? 8 : 0); }
// B-fragment (32x16 bf16) k index for reg r, lane-half h
__device__ __forceinline__ int kidxB(int r, int h) { return 2*r + 8*h + (r >= 4 ? 8 : 0); }

union BFPack { unsigned int u; __bf16 h[2]; };

// ---------------------------------------------------------------------------
// elementwise helpers
// ---------------------------------------------------------------------------
// vectorized: processes 4 floats per thread (n must be a multiple of 4)
__global__ void k_f32_to_bf16(const float* __restrict__ in, __bf16* __restrict__ out, size_t n4) {
    size_t i = (size_t)blockIdx.x * 256 + threadIdx.x;
    if (i < n4) {
        float4 v = ((const float4*)in)[i];
        BFPack p0, p1;
        p0.h[0] = f2bf(v.x); p0.h[1] = f2bf(v.y);
        p1.h[0] = f2bf(v.z); p1.h[1] = f2bf(v.w);
        ((uint2*)out)[i] = make_uint2(p0.u, p1.u);
    }
}

// tiled transpose + convert: Wt[n][k] = bf16(W[k][n]); W is [K,N] f32.
// block 32x8, tile 32x32, both global sides coalesced.
__global__ __launch_bounds__(256)
void k_cvt_transpose(const float* __restrict__ W, __bf16* __restrict__ Wt, int K, int N) {
    __shared__ float tile[32][33];
    int n0 = blockIdx.x * 32, k0 = blockIdx.y * 32;
    int tx = threadIdx.x, ty = threadIdx.y;
    #pragma unroll
    for (int j = 0; j < 4; ++j)
        tile[ty + j * 8][tx] = W[(size_t)(k0 + ty + j * 8) * N + n0 + tx];
    __syncthreads();
    #pragma unroll
    for (int j = 0; j < 4; ++j)
        Wt[(size_t)(n0 + ty + j * 8) * K + k0 + tx] = f2bf(tile[tx][ty + j * 8]);
}

__global__ void k_zero_f32(float* __restrict__ p, int n) {
    int i = blockIdx.x * 256 + threadIdx.x;
    if (i < n) p[i] = 0.0f;
}

// RoPE: read f32 Q/K, rotate, write bf16
__global__ void k_rope(const float* __restrict__ Qf, const float* __restrict__ Kf,
                       const int* __restrict__ pos,
                       __bf16* __restrict__ Qb, __bf16* __restrict__ Kb) {
    size_t gid = (size_t)blockIdx.x * 256 + threadIdx.x;
    size_t total = (size_t)SEQ * (NH + NKV) * 64;
    if (gid >= total) return;
    int j = (int)(gid & 63);
    size_t rest = gid >> 6;
    int hidx = (int)(rest % (NH + NKV));
    int t    = (int)(rest / (NH + NKV));
    float p = (float)pos[t];
    float ang = p * __powf(ROPE_THETA, -(float)(2 * j) * (1.0f / (float)HD));
    float sn, cs;
    __sincosf(ang, &sn, &cs);
    const float* src; __bf16* dst;
    if (hidx < NH) { src = Qf + (size_t)t * HID + hidx * HD; dst = Qb + (size_t)t * HID + hidx * HD; }
    else { int h2 = hidx - NH; src = Kf + (size_t)t * KVW + h2 * HD; dst = Kb + (size_t)t * KVW + h2 * HD; }
    float x1 = src[j], x2 = src[j + 64];
    dst[j]      = f2bf(x1 * cs - x2 * sn);
    dst[j + 64] = f2bf(x2 * cs + x1 * sn);
}

// V transpose per kv head: Vt[kv][d][t] = V[t][kv*128+d], bf16
__global__ void k_vtrans(const float* __restrict__ Vf, __bf16* __restrict__ Vt) {
    size_t gid = (size_t)blockIdx.x * 256 + threadIdx.x;
    if (gid >= (size_t)NKV * HD * SEQ) return;
    int t = (int)(gid & (SEQ - 1));
    size_t rest = gid >> 11;
    int d  = (int)(rest & (HD - 1));
    int kv = (int)(rest >> 7);
    Vt[gid] = f2bf(Vf[(size_t)t * KVW + kv * HD + d]);
}

// ---------------------------------------------------------------------------
// bf16 GEMM: C[M,N] f32 = A[M,K]bf16 @ Bt[N,K]bf16^T  (both row-major)
// Block tile 128x128, BK=32. 8 waves as 2(m) x 4(n); each wave owns 64x32
// (4x2 WMMA tiles => 8 v_wmma per K step). Both tiles staged into LDS with
// gfx1250 async loads (ASYNCcnt); all fragment reads are ds_load_b128.
// ---------------------------------------------------------------------------
__global__ __launch_bounds__(256)
void k_gemm_bf16(const __bf16* __restrict__ A, const __bf16* __restrict__ Bt,
                 float* __restrict__ C, int M, int N, int K) {
    // [0, 8192):    A tile 128 m-rows x 32 k, row-major
    // [8192,16384): B tile 128 n-rows x 32 k, row-major
    __shared__ __align__(16) char smem[16384];
    const unsigned smbase = (unsigned)(uintptr_t)(void*)smem;  // LDS aperture: low 32 bits = LDS addr
    const int tid = threadIdx.x, lane = tid & 31, w = tid >> 5;
    const int hl = lane >> 4, ln = lane & 15;
    const int m0 = blockIdx.y * 128, n0 = blockIdx.x * 128;
    const int wm = (w & 1) * 64, wn = (w >> 1) * 32;
    v8f acc[4][2] = {};

    for (int kb = 0; kb < K; kb += 32) {
        // stage tiles: 2 x b128 for A + 2 x b128 for B per thread, async to LDS
        #pragma unroll
        for (int j = 0; j < 2; ++j) {
            int idx = tid + j * 256;
            int row = idx >> 2, part = idx & 3;              // 128 rows x 4 x 16B each
            unsigned aoff = smbase + (unsigned)(idx * 16);
            unsigned long long ga =
                (unsigned long long)(uintptr_t)(A + (size_t)(m0 + row) * K + kb + part * 8);
            asm volatile("global_load_async_to_lds_b128 %0, %1, off"
                         :: "v"(aoff), "v"(ga) : "memory");
            unsigned boff = smbase + 8192u + (unsigned)(idx * 16);
            unsigned long long gb =
                (unsigned long long)(uintptr_t)(Bt + (size_t)(n0 + row) * K + kb + part * 8);
            asm volatile("global_load_async_to_lds_b128 %0, %1, off"
                         :: "v"(boff), "v"(gb) : "memory");
            if (kb + 32 < K) {                               // prefetch next K tile into L2
                __builtin_prefetch(A + (size_t)(m0 + row) * K + kb + 32 + part * 8, 0, 3);
                __builtin_prefetch(Bt + (size_t)(n0 + row) * K + kb + 32 + part * 8, 0, 3);
            }
        }
        asm volatile("s_wait_asynccnt 0" ::: "memory");      // our LDS writes landed
        __syncthreads();                                     // everyone's writes visible

        v16bf af[4], bfm[2];
        const unsigned* Au = (const unsigned*)smem;
        #pragma unroll
        for (int tm = 0; tm < 4; ++tm) {
            int mrow = wm + tm * 16 + ln;
            v8u au;
            #pragma unroll
            for (int r = 0; r < 8; ++r) au[r] = Au[mrow * 16 + (kidxA(r, hl) >> 1)];
            af[tm] = __builtin_bit_cast(v16bf, au);
        }
        const unsigned* Bu = (const unsigned*)(smem + 8192);
        #pragma unroll
        for (int tn = 0; tn < 2; ++tn) {
            int ncol = wn + tn * 16 + ln;
            v8u bu;
            #pragma unroll
            for (int r = 0; r < 8; ++r) bu[r] = Bu[ncol * 16 + (kidxB(r, hl) >> 1)];
            bfm[tn] = __builtin_bit_cast(v16bf, bu);
        }
        #pragma unroll
        for (int tm = 0; tm < 4; ++tm)
            #pragma unroll
            for (int tn = 0; tn < 2; ++tn)
                acc[tm][tn] = __builtin_amdgcn_wmma_f32_16x16x32_bf16(
                    false, af[tm], false, bfm[tn], (short)0, acc[tm][tn], false, false);
        __syncthreads();                                     // done reading before next stage
    }
    #pragma unroll
    for (int tm = 0; tm < 4; ++tm)
        #pragma unroll
        for (int tn = 0; tn < 2; ++tn)
            #pragma unroll
            for (int r = 0; r < 8; ++r) {
                int m = m0 + wm + tm * 16 + r + 8 * hl;
                int n = n0 + wn + tn * 16 + ln;
                C[(size_t)m * N + n] = acc[tm][tn][r];
            }
}

// ---------------------------------------------------------------------------
// Attention pass A: per (head, 16-row q block) one wave computes row max + sumexp
// ---------------------------------------------------------------------------
__global__ __launch_bounds__(32)
void k_attn_ml(const __bf16* __restrict__ Qb, const __bf16* __restrict__ Kb,
               float2* __restrict__ ml) {
    const int lane = threadIdx.x;
    const int hl = lane >> 4, ln = lane & 15;
    const int qb = blockIdx.x, head = blockIdx.y;
    const int q0 = qb * 16;
    const int kv = head >> 2;
    const __bf16* Qh = Qb + (size_t)head * HD;
    const __bf16* Kh = Kb + (size_t)kv * HD;

    v16bf qf[4];
    {
        int m = q0 + ln;
        #pragma unroll
        for (int kk = 0; kk < 4; ++kk) {
            v8u au;
            #pragma unroll
            for (int r = 0; r < 8; ++r)
                au[r] = *(const unsigned int*)(Qh + (size_t)m * HID + kk * 32 + kidxA(r, hl));
            qf[kk] = __builtin_bit_cast(v16bf, au);
        }
    }
    float mrun[8], lrun[8];
    #pragma unroll
    for (int r = 0; r < 8; ++r) { mrun[r] = -1e30f; lrun[r] = 0.0f; }

    for (int kb = 0; kb <= qb; ++kb) {
        v8f s = {};
        int key = kb * 16 + ln;
        #pragma unroll
        for (int kk = 0; kk < 4; ++kk) {
            v8u bu;
            #pragma unroll
            for (int r = 0; r < 8; ++r)
                bu[r] = *(const unsigned int*)(Kh + (size_t)key * KVW + kk * 32 + kidxB(r, hl));
            v16bf bfm = __builtin_bit_cast(v16bf, bu);
            s = __builtin_amdgcn_wmma_f32_16x16x32_bf16(false, qf[kk], false, bfm, (short)0, s, false, false);
        }
        #pragma unroll
        for (int r = 0; r < 8; ++r) {
            int mg = q0 + r + 8 * hl;
            float sv = s[r] * QK_SCALE;
            if (key > mg) sv = -1e30f;                 // causal mask
            float bm = sv;
            bm = fmaxf(bm, __shfl_xor(bm, 1, 32));
            bm = fmaxf(bm, __shfl_xor(bm, 2, 32));
            bm = fmaxf(bm, __shfl_xor(bm, 4, 32));
            bm = fmaxf(bm, __shfl_xor(bm, 8, 32));
            float nm = fmaxf(mrun[r], bm);
            float e = __expf(sv - nm);
            e += __shfl_xor(e, 1, 32); e += __shfl_xor(e, 2, 32);
            e += __shfl_xor(e, 4, 32); e += __shfl_xor(e, 8, 32);
            lrun[r] = lrun[r] * __expf(mrun[r] - nm) + e;
            mrun[r] = nm;
        }
    }
    if (ln == 0) {
        #pragma unroll
        for (int r = 0; r < 8; ++r) {
            int row = q0 + r + 8 * hl;
            ml[(size_t)head * SEQ + row] = make_float2(mrun[r], lrun[r]);
        }
    }
}

// ---------------------------------------------------------------------------
// Attention pass B: recompute scores, normalize, P@V via WMMA, hh atomics
// ---------------------------------------------------------------------------
__global__ __launch_bounds__(32)
void k_attn_out(const __bf16* __restrict__ Qb, const __bf16* __restrict__ Kb,
                const __bf16* __restrict__ Vt, const float2* __restrict__ ml,
                float* __restrict__ Of, float* __restrict__ hh) {
    __shared__ __align__(4) __bf16 Pld[16 * 32];
    const int lane = threadIdx.x;
    const int hl = lane >> 4, ln = lane & 15;
    const int qb = blockIdx.x, head = blockIdx.y;
    const int q0 = qb * 16;
    const int kv = head >> 2;
    const __bf16* Qh = Qb + (size_t)head * HD;
    const __bf16* Kh = Kb + (size_t)kv * HD;
    const __bf16* Vh = Vt + (size_t)kv * HD * SEQ;

    v16bf qf[4];
    {
        int m = q0 + ln;
        #pragma unroll
        for (int kk = 0; kk < 4; ++kk) {
            v8u au;
            #pragma unroll
            for (int r = 0; r < 8; ++r)
                au[r] = *(const unsigned int*)(Qh + (size_t)m * HID + kk * 32 + kidxA(r, hl));
            qf[kk] = __builtin_bit_cast(v16bf, au);
        }
    }
    float mrow[8], linv[8];
    #pragma unroll
    for (int r = 0; r < 8; ++r) {
        float2 v = ml[(size_t)head * SEQ + q0 + r + 8 * hl];
        mrow[r] = v.x; linv[r] = 1.0f / v.y;
    }
    v8f acc[8] = {};
    const int nch = qb / 2 + 1;                       // 32-key chunks
    for (int ch = 0; ch < nch; ++ch) {
        #pragma unroll
        for (int t = 0; t < 2; ++t) {
            int kb = ch * 2 + t;
            if (kb <= qb) {                           // uniform branch: EXEC stays full
                v8f s = {};
                int key = kb * 16 + ln;
                #pragma unroll
                for (int kk = 0; kk < 4; ++kk) {
                    v8u bu;
                    #pragma unroll
                    for (int r = 0; r < 8; ++r)
                        bu[r] = *(const unsigned int*)(Kh + (size_t)key * KVW + kk * 32 + kidxB(r, hl));
                    v16bf bfm = __builtin_bit_cast(v16bf, bu);
                    s = __builtin_amdgcn_wmma_f32_16x16x32_bf16(false, qf[kk], false, bfm, (short)0, s, false, false);
                }
                float csum = 0.0f;
                #pragma unroll
                for (int r = 0; r < 8; ++r) {
                    int mg = q0 + r + 8 * hl;
                    float sv = s[r] * QK_SCALE;
                    if (key > mg) sv = -1e30f;
                    float p = __expf(sv - mrow[r]) * linv[r];
                    Pld[(r + 8 * hl) * 32 + t * 16 + ln] = f2bf(p);
                    csum += p;
                }
                csum += __shfl_xor(csum, 16, 32);     // combine row halves per column
                if (lane < 16) atomicAdd(&hh[(size_t)kv * SEQ + kb * 16 + ln], csum);
            } else {
                #pragma unroll
                for (int r = 0; r < 8; ++r) Pld[(r + 8 * hl) * 32 + t * 16 + ln] = f2bf(0.0f);
            }
        }
        __syncthreads();
        v16bf pf;
        {
            v8u pu;
            #pragma unroll
            for (int r = 0; r < 8; ++r)
                pu[r] = *(const unsigned int*)(&Pld[ln * 32 + kidxA(r, hl)]);
            pf = __builtin_bit_cast(v16bf, pu);
        }
        #pragma unroll
        for (int d = 0; d < 8; ++d) {
            v8u bu;
            #pragma unroll
            for (int r = 0; r < 8; ++r)
                bu[r] = *(const unsigned int*)(Vh + (size_t)(d * 16 + ln) * SEQ + ch * 32 + kidxB(r, hl));
            v16bf bfm = __builtin_bit_cast(v16bf, bu);
            acc[d] = __builtin_amdgcn_wmma_f32_16x16x32_bf16(false, pf, false, bfm, (short)0, acc[d], false, false);
        }
        __syncthreads();
    }
    #pragma unroll
    for (int d = 0; d < 8; ++d)
        #pragma unroll
        for (int r = 0; r < 8; ++r) {
            int row = q0 + r + 8 * hl;
            int col = head * HD + d * 16 + ln;
            Of[(size_t)row * HID + col] = acc[d][r];
        }
}

// ---------------------------------------------------------------------------
// launch
// ---------------------------------------------------------------------------
extern "C" void kernel_launch(void* const* d_in, const int* in_sizes, int n_in,
                              void* d_out, int out_size, void* d_ws, size_t ws_size,
                              hipStream_t stream) {
    (void)in_sizes; (void)n_in; (void)out_size; (void)ws_size;
    const float* X   = (const float*)d_in[0];
    const int*   pos = (const int*)d_in[1];
    const float* Wq  = (const float*)d_in[2];
    const float* Wk  = (const float*)d_in[3];
    const float* Wv  = (const float*)d_in[4];
    const float* Wo  = (const float*)d_in[5];

    char* ws = (char*)d_ws;
    __bf16* Xb  = (__bf16*)(ws + OFF_XB);
    __bf16* Wqt = (__bf16*)(ws + OFF_WQB);
    __bf16* Wkt = (__bf16*)(ws + OFF_WKB);
    __bf16* Wvt = (__bf16*)(ws + OFF_WVB);
    __bf16* Wot = (__bf16*)(ws + OFF_WOB);
    float*  Qf  = (float*)(ws + OFF_QF);
    float*  Kf  = (float*)(ws + OFF_KF);
    float*  Vf  = (float*)(ws + OFF_VF);
    __bf16* Qb  = (__bf16*)(ws + OFF_QB);
    __bf16* Kb  = (__bf16*)(ws + OFF_KB);
    __bf16* Vt  = (__bf16*)(ws + OFF_VT);
    float2* ml  = (float2*)(ws + OFF_ML);
    float*  Of  = (float*)(ws + OFF_OF);
    __bf16* Ob  = (__bf16*)(ws + OFF_OB);

    float* out = (float*)d_out;                        // [2048,4096]
    float* hh  = out + (size_t)SEQ * HID;              // [8,2048]

    auto cvt = [&](const float* src, __bf16* dst, size_t n) {
        size_t n4 = n / 4;
        k_f32_to_bf16<<<(unsigned)((n4 + 255) / 256), 256, 0, stream>>>(src, dst, n4);
    };
    auto cvtT = [&](const float* srcKN, __bf16* dstNK, int K, int N) {
        k_cvt_transpose<<<dim3(N / 32, K / 32), dim3(32, 8), 0, stream>>>(srcKN, dstNK, K, N);
    };
    cvt(X, Xb, (size_t)SEQ * HID);
    cvtT(Wq, Wqt, HID, HID);   // Wq [H,4096] -> Wq^T [4096,H]
    cvtT(Wk, Wkt, HID, KVW);   // Wk [H,1024] -> Wk^T [1024,H]
    cvtT(Wv, Wvt, HID, KVW);
    cvtT(Wo, Wot, HID, HID);

    // QKV projections (N divisible by 128 in all cases)
    k_gemm_bf16<<<dim3(HID / 128, SEQ / 128), 256, 0, stream>>>(Xb, Wqt, Qf, SEQ, HID, HID);
    k_gemm_bf16<<<dim3(KVW / 128, SEQ / 128), 256, 0, stream>>>(Xb, Wkt, Kf, SEQ, KVW, HID);
    k_gemm_bf16<<<dim3(KVW / 128, SEQ / 128), 256, 0, stream>>>(Xb, Wvt, Vf, SEQ, KVW, HID);

    // RoPE (Q,K -> bf16), V transpose (-> bf16)
    {
        size_t n = (size_t)SEQ * (NH + NKV) * 64;
        k_rope<<<(unsigned)((n + 255) / 256), 256, 0, stream>>>(Qf, Kf, pos, Qb, Kb);
        size_t nv = (size_t)NKV * HD * SEQ;
        k_vtrans<<<(unsigned)((nv + 255) / 256), 256, 0, stream>>>(Vf, Vt);
    }

    // hh accumulators must start at zero every call
    k_zero_f32<<<(NKV * SEQ + 255) / 256, 256, 0, stream>>>(hh, NKV * SEQ);

    // attention: pass A (m,l), pass B (output + hh)
    k_attn_ml<<<dim3(SEQ / 16, NH), 32, 0, stream>>>(Qb, Kb, ml);
    k_attn_out<<<dim3(SEQ / 16, NH), 32, 0, stream>>>(Qb, Kb, Vt, ml, Of, hh);

    // output projection
    cvt(Of, Ob, (size_t)SEQ * HID);
    k_gemm_bf16<<<dim3(HID / 128, SEQ / 128), 256, 0, stream>>>(Ob, Wot, out, SEQ, HID, HID);
}
